// MetaController_53077205844185
// MI455X (gfx1250) — matrix operand order repaired and checked by reading, not compiled, not run on hardware
//
#include <hip/hip_runtime.h>
#include <hip/hip_bf16.h>
#include <math.h>

// ---------------------------------------------------------------------------
// MetaController for MI455X (gfx1250).  bf16 WMMA for all parallel GEMMs,
// persistent VALU blocks for the serial GRU recurrences, register-fused
// w1*s contraction so the 537MB dec_W2 output is never materialized.
// ---------------------------------------------------------------------------

typedef __bf16 bf16;
typedef __attribute__((ext_vector_type(16))) __bf16 v16bf;
typedef __attribute__((ext_vector_type(8)))  float  v8f;
typedef __attribute__((ext_vector_type(4)))  float  f32x4;

#define BB   4
#define NN   2048
#define DD   512
#define ROWS (BB*NN)      // 8192
#define H3   1536
#define DH   1024
#define DO   16384
#define RR   16

__device__ __forceinline__ float sigmoidf_(float x) { return 1.0f / (1.0f + __expf(-x)); }

// ---- WMMA fragment loaders (layouts per CDNA5 ISA 7.12.2) -------------------
// A: 16x32 bf16 tile, row-major source with row stride lda (elements).
__device__ __forceinline__ v16bf load_a_frag(const bf16* __restrict__ A, int lda,
                                             int m0, int k0, int lane) {
  const int r  = lane & 15;
  const int hk = lane >> 4;              // K-half select
  const bf16* p = A + (size_t)(m0 + r) * lda + k0 + hk * 8;
  union { v16bf v; f32x4 q[2]; } u;
  u.q[0] = *(const f32x4*)(p);           // elems 0..7  : K = k0+hk*8   + 0..7
  u.q[1] = *(const f32x4*)(p + 16);      // elems 8..15 : K = k0+16+hk*8+ 0..7
  return u.v;
}
// B: 32x16 bf16 tile == 16 rows of W (N,K row-major); lane holds one column.
__device__ __forceinline__ v16bf load_b_frag(const bf16* __restrict__ W, int ldw,
                                             int n0, int k0, int lane) {
  const int c  = lane & 15;
  const int kh = lane >> 4;
  const bf16* p = W + (size_t)(n0 + c) * ldw + k0 + kh * 8;
  union { v16bf v; f32x4 q[2]; } u;
  u.q[0] = *(const f32x4*)(p);           // K = k0+kh*8 + 0..7
  u.q[1] = *(const f32x4*)(p + 16);      // K = k0+16+kh*8 + 0..7
  return u.v;
}

// ---- Generic GEMM:  C(MxN) = A(MxK) * W(NxK)^T + bias  ----------------------
// EPI: 0 = store f32, 1 = sigmoid -> f32, 2 = silu -> bf16
template <int EPI>
__global__ void gemm_bf16_wmma(const bf16* __restrict__ A,
                               const bf16* __restrict__ W,
                               const float* __restrict__ bias,
                               float* __restrict__ Cf, bf16* __restrict__ Cb,
                               int N, int K, int ldc) {
  const int lane = threadIdx.x & 31;
  const int wave = threadIdx.x >> 5;
  const int m0   = blockIdx.x * 16;
  const int n0   = (blockIdx.y * (blockDim.x >> 5) + wave) * 64;
  if (n0 >= N) return;

  v8f acc[4];
#pragma unroll
  for (int t = 0; t < 4; t++) {
    float bv = 0.0f;
    if (bias && (n0 + 16 * t) < N) bv = bias[n0 + 16 * t + (lane & 15)];
#pragma unroll
    for (int i = 0; i < 8; i++) acc[t][i] = bv;
  }

  for (int k0 = 0; k0 < K; k0 += 32) {
    v16bf a = load_a_frag(A, K, m0, k0, lane);
#pragma unroll
    for (int t = 0; t < 4; t++) {
      if (n0 + 16 * t >= N) break;
      v16bf b = load_b_frag(W, K, n0 + 16 * t, k0, lane);
      acc[t] = __builtin_amdgcn_wmma_f32_16x16x32_bf16(false, a, false, b,
                                                       (short)0, acc[t], false, false);
    }
  }

#pragma unroll
  for (int t = 0; t < 4; t++) {
    const int nb = n0 + 16 * t;
    if (nb >= N) break;
    const int col = nb + (lane & 15);
#pragma unroll
    for (int i = 0; i < 8; i++) {
      const int row = m0 + (lane >> 4) * 8 + i;
      float v = acc[t][i];
      if (EPI == 1) v = sigmoidf_(v);
      if (EPI == 2) v = v * sigmoidf_(v);
      const size_t off = (size_t)row * ldc + col;
      if (EPI == 2) Cb[off] = (bf16)v;
      else          Cf[off] = v;
    }
  }
}

// ---- fp32 -> bf16 bulk convert ---------------------------------------------
__global__ void cvt_kernel(const float* __restrict__ src, bf16* __restrict__ dst, size_t n) {
  size_t i = (size_t)blockIdx.x * blockDim.x + threadIdx.x;
  if (i < n) dst[i] = (bf16)src[i];
}

// ---- transpose W_hh (1536x512 f32) -> WT (512x1536 bf16), k-major -----------
__global__ void whht_kernel(const float* __restrict__ W, bf16* __restrict__ T) {
  int i = blockIdx.x * blockDim.x + threadIdx.x;
  if (i >= H3 * DD) return;
  int g = i >> 9, k = i & 511;
  T[(size_t)k * H3 + g] = (bf16)W[i];
}

// ---- pre-reduce the w2 half of dec_W2:  W2s[r,k] = sum_d W2[8192+d*16+r, k] -
__global__ void w2s_kernel(const float* __restrict__ W2, const float* __restrict__ b2,
                           bf16* __restrict__ W2s, float* __restrict__ b2s) {
  int i = blockIdx.x * blockDim.x + threadIdx.x;
  if (i >= RR * DH) return;
  int k = i & (DH - 1), r = i >> 10;
  float s = 0.0f;
  for (int d = 0; d < DD; d++) s += W2[((size_t)(DD * RR + d * RR + r)) * DH + k];
  W2s[(size_t)r * DH + k] = (bf16)s;
  if (k == 0) {
    float bs = 0.0f;
    for (int d = 0; d < DD; d++) bs += b2[DD * RR + d * RR + r];
    b2s[r] = bs;
  }
}

// ---- persistent GRU recurrence: one block per (batch, gru) ------------------
__global__ __launch_bounds__(512) void gru_kernel(
    const float* __restrict__ xp_ap, const float* __restrict__ xp_su,
    const bf16* __restrict__ WT_ap, const bf16* __restrict__ WT_su,
    const float* __restrict__ bhh_ap, const float* __restrict__ bhh_su,
    bf16* __restrict__ hs_ap, bf16* __restrict__ hs_su) {
  const int b = blockIdx.x;
  const int g = blockIdx.y;
  const float* xp  = g ? xp_su  : xp_ap;
  const bf16*  WT  = g ? WT_su  : WT_ap;
  const float* bhh = g ? bhh_su : bhh_ap;
  bf16*        hs  = g ? hs_su  : hs_ap;
  const int j = threadIdx.x;

  __shared__ float hsh[DD];
  hsh[j] = 0.0f;
  const float br = bhh[j], bz = bhh[DD + j], bn_ = bhh[2 * DD + j];
  __syncthreads();

  for (int t = 0; t < NN; t++) {
    float ar = br, az = bz, an = bn_;
#pragma unroll 4
    for (int k = 0; k < DD; k++) {
      const float hk = hsh[k];
      const bf16* wr = WT + (size_t)k * H3;
      ar += hk * (float)wr[j];
      az += hk * (float)wr[DD + j];
      an += hk * (float)wr[2 * DD + j];
    }
    const size_t row = (size_t)b * NN + t;
    const float* xr = xp + row * H3;
    const float r  = sigmoidf_(xr[j] + ar);
    const float z  = sigmoidf_(xr[DD + j] + az);
    const float nn = tanhf(xr[2 * DD + j] + r * an);
    const float hnew = (1.0f - z) * nn + z * hsh[j];
    __syncthreads();
    hsh[j] = hnew;
    hs[row * DD + j] = (bf16)hnew;
    __syncthreads();
  }
}

// ---- sampled_action + kl ----------------------------------------------------
__global__ void sample_kernel(const float* __restrict__ act, const float* __restrict__ eps,
                              float* __restrict__ sampled, float* __restrict__ kl) {
  size_t idx = (size_t)blockIdx.x * blockDim.x + threadIdx.x;
  if (idx == 0) *kl = 0.0f;
  if (idx >= (size_t)ROWS * DD) return;
  size_t row = idx >> 9;
  int d = idx & (DD - 1);
  float mean = act[row * (2 * DD) + 2 * d];
  float lv   = act[row * (2 * DD) + 2 * d + 1];
  sampled[idx] = mean + __expf(0.5f * lv) * eps[idx];
}

// ---- linear-recurrence scan over N: h = beta*h + sampled*(1-beta) -----------
__global__ __launch_bounds__(512) void scan_kernel(const float* __restrict__ sampled,
                                                   const float* __restrict__ beta,
                                                   float* __restrict__ gated,
                                                   bf16* __restrict__ gated_bf) {
  const int b = blockIdx.x, d = threadIdx.x;
  float h = 0.0f;
  for (int n = 0; n < NN; n++) {
    const size_t idx = ((size_t)b * NN + n) * DD + d;
    const float bt = beta[idx];
    h = bt * h + sampled[idx] * (1.0f - bt);
    gated[idx]    = h;
    gated_bf[idx] = (bf16)h;
  }
}

// ---- fused w1-half of dec_W2 GEMM + contraction with s ----------------------
// t[row,d] = sum_r (hdec[row,:]·W2[d*16+r,:] + b2[d*16+r]) * s[row,r]
__global__ __launch_bounds__(256) void wsum_kernel(const bf16* __restrict__ Hd,
                                                   const bf16* __restrict__ W2,
                                                   const float* __restrict__ b2,
                                                   const float* __restrict__ S,
                                                   float* __restrict__ T) {
  const int lane = threadIdx.x & 31, wave = threadIdx.x >> 5;
  const int m0 = blockIdx.x * 16;
  const int dbase = blockIdx.y * 32 + wave * 4;

  __shared__ float ssh[16][16];
  { int m = threadIdx.x >> 4, r = threadIdx.x & 15;
    ssh[m][r] = S[(size_t)(m0 + m) * RR + r]; }
  __syncthreads();

  v8f acc[4];
#pragma unroll
  for (int dd = 0; dd < 4; dd++) {
    const float bv = b2[(dbase + dd) * RR + (lane & 15)];
#pragma unroll
    for (int i = 0; i < 8; i++) acc[dd][i] = bv;
  }

  for (int k0 = 0; k0 < DH; k0 += 32) {
    v16bf a = load_a_frag(Hd, DH, m0, k0, lane);
#pragma unroll
    for (int dd = 0; dd < 4; dd++) {
      v16bf b = load_b_frag(W2, DH, (dbase + dd) * RR, k0, lane);
      acc[dd] = __builtin_amdgcn_wmma_f32_16x16x32_bf16(false, a, false, b,
                                                        (short)0, acc[dd], false, false);
    }
  }

  const int r = lane & 15, half = lane >> 4;
#pragma unroll
  for (int dd = 0; dd < 4; dd++) {
#pragma unroll
    for (int i = 0; i < 8; i++) {
      const int m = half * 8 + i;
      float p = acc[dd][i] * ssh[m][r];
      p += __shfl_xor(p, 1);
      p += __shfl_xor(p, 2);
      p += __shfl_xor(p, 4);
      p += __shfl_xor(p, 8);          // sum over r within the 16-lane half
      if (r == i) T[(size_t)(m0 + m) * DD + (dbase + dd)] = p;
    }
  }
}

// ---- final: modified = residual + gated * t ---------------------------------
__global__ void final_kernel(const float* __restrict__ residual,
                             const float* __restrict__ gated,
                             const float* __restrict__ t,
                             float* __restrict__ modified) {
  size_t idx = (size_t)blockIdx.x * blockDim.x + threadIdx.x;
  if (idx >= (size_t)ROWS * DD) return;
  modified[idx] = residual[idx] + gated[idx] * t[idx];
}

// ---------------------------------------------------------------------------
extern "C" void kernel_launch(void* const* d_in, const int* in_sizes, int n_in,
                              void* d_out, int out_size, void* d_ws, size_t ws_size,
                              hipStream_t stream) {
  (void)in_sizes; (void)n_in; (void)out_size; (void)ws_size;
  const float* residual = (const float*)d_in[0];
  const float* eps      = (const float*)d_in[1];
  const float* ap_W_ih  = (const float*)d_in[2];
  const float* ap_W_hh  = (const float*)d_in[3];
  const float* ap_b_ih  = (const float*)d_in[4];
  const float* ap_b_hh  = (const float*)d_in[5];
  const float* su_W_ih  = (const float*)d_in[6];
  const float* su_W_hh  = (const float*)d_in[7];
  const float* su_b_ih  = (const float*)d_in[8];
  const float* su_b_hh  = (const float*)d_in[9];
  const float* readout_W = (const float*)d_in[10];
  const float* readout_b = (const float*)d_in[11];
  const float* beta_W   = (const float*)d_in[12];
  const float* dec_W1   = (const float*)d_in[13];
  const float* dec_b1   = (const float*)d_in[14];
  const float* dec_W2   = (const float*)d_in[15];
  const float* dec_b2   = (const float*)d_in[16];

  float* out = (float*)d_out;
  float* out_modified = out;                                 // 4,194,304
  float* out_action   = out + (size_t)ROWS * DD;             // 8,388,608 (B,N,D,2)
  float* out_sampled  = out + (size_t)ROWS * DD * 3;         // 4,194,304
  float* out_kl       = out + (size_t)ROWS * DD * 4;         // scalar

  // ---- workspace carve-up --------------------------------------------------
  char* base = (char*)d_ws;
  size_t o = 0;
  auto alloc = [&](size_t bytes) -> char* {
    char* p = base + o;
    o += (bytes + 255) & ~(size_t)255;
    return p;
  };
  float* xp_ap   = (float*)alloc((size_t)ROWS * H3 * 4);
  float* xp_su   = (float*)alloc((size_t)ROWS * H3 * 4);
  bf16*  x_bf    = (bf16*) alloc((size_t)ROWS * DD * 2);
  bf16*  wih_ap  = (bf16*) alloc((size_t)H3 * DD * 2);
  bf16*  wih_su  = (bf16*) alloc((size_t)H3 * DD * 2);
  bf16*  wt_ap   = (bf16*) alloc((size_t)DD * H3 * 2);
  bf16*  wt_su   = (bf16*) alloc((size_t)DD * H3 * 2);
  bf16*  rout_bf = (bf16*) alloc((size_t)(2 * DD) * DD * 2);
  bf16*  betw_bf = (bf16*) alloc((size_t)DD * DD * 2);
  bf16*  w1_bf   = (bf16*) alloc((size_t)DH * DD * 2);
  bf16*  w2_bf   = (bf16*) alloc((size_t)DO * DH * 2);
  bf16*  w2s_bf  = (bf16*) alloc((size_t)RR * DH * 2);
  float* b2s     = (float*)alloc(RR * 4);
  bf16*  h_ap_bf = (bf16*) alloc((size_t)ROWS * DD * 2);
  bf16*  h_su_bf = (bf16*) alloc((size_t)ROWS * DD * 2);
  float* beta_ws = (float*)alloc((size_t)ROWS * DD * 4);
  float* gated   = (float*)alloc((size_t)ROWS * DD * 4);
  bf16*  gated_bf= (bf16*) alloc((size_t)ROWS * DD * 2);
  bf16*  hdec_bf = (bf16*) alloc((size_t)ROWS * DH * 2);
  float* s_ws    = (float*)alloc((size_t)ROWS * RR * 4);
  float* t_ws    = (float*)alloc((size_t)ROWS * DD * 4);

  auto cvt = [&](const float* src, bf16* dst, size_t n) {
    cvt_kernel<<<(unsigned)((n + 255) / 256), 256, 0, stream>>>(src, dst, n);
  };

  // ---- stage 0: precision prep --------------------------------------------
  cvt(residual,  x_bf,    (size_t)ROWS * DD);
  cvt(ap_W_ih,   wih_ap,  (size_t)H3 * DD);
  cvt(su_W_ih,   wih_su,  (size_t)H3 * DD);
  cvt(readout_W, rout_bf, (size_t)(2 * DD) * DD);
  cvt(beta_W,    betw_bf, (size_t)DD * DD);
  cvt(dec_W1,    w1_bf,   (size_t)DH * DD);
  cvt(dec_W2,    w2_bf,   (size_t)DO * DH);
  whht_kernel<<<(H3 * DD + 255) / 256, 256, 0, stream>>>(ap_W_hh, wt_ap);
  whht_kernel<<<(H3 * DD + 255) / 256, 256, 0, stream>>>(su_W_hh, wt_su);
  w2s_kernel<<<(RR * DH + 255) / 256, 256, 0, stream>>>(dec_W2, dec_b2, w2s_bf, b2s);

  // ---- stage 1: input projections (WMMA) ----------------------------------
  gemm_bf16_wmma<0><<<dim3(ROWS / 16, 3), 256, 0, stream>>>(
      x_bf, wih_ap, ap_b_ih, xp_ap, nullptr, H3, DD, H3);
  gemm_bf16_wmma<0><<<dim3(ROWS / 16, 3), 256, 0, stream>>>(
      x_bf, wih_su, su_b_ih, xp_su, nullptr, H3, DD, H3);

  // ---- stage 2: both GRU recurrences (persistent, latency-bound) ----------
  gru_kernel<<<dim3(BB, 2), 512, 0, stream>>>(xp_ap, xp_su, wt_ap, wt_su,
                                              ap_b_hh, su_b_hh, h_ap_bf, h_su_bf);

  // ---- stage 3: readout -> action_dist, then sample -----------------------
  gemm_bf16_wmma<0><<<dim3(ROWS / 16, 2), 256, 0, stream>>>(
      h_ap_bf, rout_bf, readout_b, out_action, nullptr, 2 * DD, DD, 2 * DD);
  sample_kernel<<<((size_t)ROWS * DD + 255) / 256, 256, 0, stream>>>(
      out_action, eps, out_sampled, out_kl);

  // ---- stage 4: switch_beta = sigmoid(su @ beta_W^T) ----------------------
  gemm_bf16_wmma<1><<<dim3(ROWS / 16, 1), 256, 0, stream>>>(
      h_su_bf, betw_bf, nullptr, beta_ws, nullptr, DD, DD, DD);

  // ---- stage 5: linear recurrence scan ------------------------------------
  scan_kernel<<<BB, 512, 0, stream>>>(out_sampled, beta_ws, gated, gated_bf);

  // ---- stage 6: hdec = silu(gated @ dec_W1^T + b1) (bf16 out) -------------
  gemm_bf16_wmma<2><<<dim3(ROWS / 16, 2), 256, 0, stream>>>(
      gated_bf, w1_bf, dec_b1, nullptr, hdec_bf, DH, DD, DH);

  // ---- stage 7: s = hdec @ W2s^T + b2s  (pre-reduced w2 half) -------------
  gemm_bf16_wmma<0><<<dim3(ROWS / 16, 1), 32, 0, stream>>>(
      hdec_bf, w2s_bf, b2s, s_ws, nullptr, RR, DH, RR);

  // ---- stage 8: fused w1 GEMM + contraction with s -> t -------------------
  wsum_kernel<<<dim3(ROWS / 16, DD / 32), 256, 0, stream>>>(
      hdec_bf, w2_bf, dec_b2, s_ws, t_ws);

  // ---- stage 9: modified = residual + gated * t ---------------------------
  final_kernel<<<((size_t)ROWS * DD + 255) / 256, 256, 0, stream>>>(
      residual, gated, t_ws, out_modified);
}
